// MultiheadedSelfAttentionModule_7739531067747
// MI455X (gfx1250) — matrix operand order repaired and checked by reading, not compiled, not run on hardware
//
#include <hip/hip_runtime.h>

// ---------------------------------------------------------------------------
// MultiheadedSelfAttention for gfx1250 (CDNA5), f16 WMMA with f32 accumulate.
//   x:      [4, 2048, 1024] f32
//   W_QKV:  [1024, 3072]    f32   (output interleaved as [h][hd][qkv3])
//   W_Out:  [1024, 1024]    f32
//   out:    [4, 2048, 1024] f32
// Stages: (1) QKV GEMM + deinterleave -> Q(scaled)/K/V^T f16 in workspace,
//         (2) flash attention via WMMA, (3) out projection GEMM.
// ---------------------------------------------------------------------------

typedef __attribute__((ext_vector_type(16))) _Float16 v16h;
typedef __attribute__((ext_vector_type(8)))  _Float16 v8h;
typedef __attribute__((ext_vector_type(8)))  float    v8f;
typedef __attribute__((ext_vector_type(4)))  float    v4f;

// Async global->LDS copy path (ASYNCcnt) when the toolchain exposes it.
#if defined(__has_builtin)
#if __has_builtin(__builtin_amdgcn_global_load_async_to_lds_b128) && \
    __has_builtin(__builtin_amdgcn_s_wait_asynccnt)
#define USE_ASYNC_LDS 1
#endif
#endif
#ifndef USE_ASYNC_LDS
#define USE_ASYNC_LDS 0
#endif

#if USE_ASYNC_LDS
// Builtin signature (from clang diagnostics): pointers to 16-byte int vectors,
// global operand in AS(1), LDS operand in AS(3).
typedef int b128_t __attribute__((vector_size(16)));
typedef __attribute__((address_space(1))) b128_t* gas_b128;
typedef __attribute__((address_space(3))) b128_t* las_b128;
static __device__ __forceinline__ void async_cp16(const void* g, void* l) {
  __builtin_amdgcn_global_load_async_to_lds_b128((gas_b128)g, (las_b128)l, 0, 0);
}
#define ASYNC_CP16(gptr, lptr) async_cp16((const void*)(gptr), (void*)(lptr))
#define ASYNC_WAIT() __builtin_amdgcn_s_wait_asynccnt(0)
#endif

static __device__ __forceinline__ v16h join8(v8h lo, v8h hi) {
  return __builtin_shufflevector(lo, hi, 0,1,2,3,4,5,6,7,8,9,10,11,12,13,14,15);
}

static __device__ __forceinline__ v8f wmma_f16(v16h a, v16h b, v8f c) {
  // D = A(16x32 f16) * B(32x16 f16) + C(16x16 f32)
  return __builtin_amdgcn_wmma_f32_16x16x32_f16(false, a, false, b, (short)0, c, false, false);
}

static __device__ __forceinline__ float rmax16(float v) {
  v = fmaxf(v, __shfl_xor(v, 1));
  v = fmaxf(v, __shfl_xor(v, 2));
  v = fmaxf(v, __shfl_xor(v, 4));
  v = fmaxf(v, __shfl_xor(v, 8));
  return v;
}
static __device__ __forceinline__ float rsum16(float v) {
  v += __shfl_xor(v, 1);
  v += __shfl_xor(v, 2);
  v += __shfl_xor(v, 4);
  v += __shfl_xor(v, 8);
  return v;
}

// A-fragment (16x32 f16, M x K): lane holds row M = lane&15.
//   lane<16: halves 0..15 = K 0..15
//   lane>=16: halves 0..15 = K 16..31
// -> two contiguous 8-half loads at K offsets kb and kb+8, kb = (lane<16)?0:16.
static __device__ __forceinline__ v16h loadA(const _Float16* row_k0, int kb) {
  v8h lo = *(const v8h*)(row_k0 + kb);
  v8h hi = *(const v8h*)(row_k0 + kb + 8);
  return join8(lo, hi);
}
// B-fragment (32x16 f16, K x N) from [n][k]-major storage: lane holds col n =
// lane&15; 16 contiguous K at kb = (lane<16)?0:16.
static __device__ __forceinline__ v16h loadB(const _Float16* n_k0, int kb) {
  v8h lo = *(const v8h*)(n_k0 + kb);
  v8h hi = *(const v8h*)(n_k0 + kb + 8);
  return join8(lo, hi);
}

// ---------------------------------------------------------------------------
// Kernel 1: QKV = x @ W_QKV, de-interleave into Q (scaled), K, V^T as f16.
//   grid (24, 64), block 256.  128x128 tile per WG, K-step 32.
// ---------------------------------------------------------------------------
__global__ __launch_bounds__(256) void qkv_kernel(
    const float* __restrict__ x, const float* __restrict__ Wqkv,
    _Float16* __restrict__ Qh, _Float16* __restrict__ Kh,
    _Float16* __restrict__ Vt)
{
  constexpr int LDA = 40, LDB = 40;
  __shared__ _Float16 lA[128 * LDA];
  __shared__ _Float16 lB[128 * LDB];

  const int tid  = threadIdx.x;
  const int lane = tid & 31, wave = tid >> 5;
  const int wm = wave >> 1, wn = wave & 1;     // 4x2 wave grid -> 32x64 per wave
  const int mBase = blockIdx.y * 128;
  const int nBase = blockIdx.x * 128;

  v8f acc[2][4];
#pragma unroll
  for (int i = 0; i < 2; ++i)
#pragma unroll
    for (int j = 0; j < 4; ++j)
      acc[i][j] = (v8f){0.f,0.f,0.f,0.f,0.f,0.f,0.f,0.f};

  const int kbA = (lane < 16) ? 0 : 16;
  const int kbB = (lane < 16) ? 0 : 16;
  const int mr  = lane & 15;

  for (int k0 = 0; k0 < 1024; k0 += 32) {
    // A tile: x[mBase..+127][k0..k0+31], f32 -> f16, [m][k]
#pragma unroll
    for (int i = 0; i < 4; ++i) {
      int idx = tid + i * 256;               // 1024 float4s
      int row = idx >> 3, kv = (idx & 7) * 4;
      v4f d = *(const v4f*)(x + (size_t)(mBase + row) * 1024 + k0 + kv);
      _Float16* dst = &lA[row * LDA + kv];
      dst[0] = (_Float16)d.x; dst[1] = (_Float16)d.y;
      dst[2] = (_Float16)d.z; dst[3] = (_Float16)d.w;
    }
    // B tile: Wqkv[k0..+31][nBase..+127], f32 -> f16, transposed to [n][k]
#pragma unroll
    for (int i = 0; i < 4; ++i) {
      int idx = tid + i * 256;               // 1024 float4s
      int r = idx >> 5, nv = (idx & 31) * 4;
      v4f d = *(const v4f*)(Wqkv + (size_t)(k0 + r) * 3072 + nBase + nv);
      lB[(nv + 0) * LDB + r] = (_Float16)d.x;
      lB[(nv + 1) * LDB + r] = (_Float16)d.y;
      lB[(nv + 2) * LDB + r] = (_Float16)d.z;
      lB[(nv + 3) * LDB + r] = (_Float16)d.w;
    }
    __syncthreads();

    v16h aF[2];
#pragma unroll
    for (int mt = 0; mt < 2; ++mt)
      aF[mt] = loadA(&lA[(wm * 32 + mt * 16 + mr) * LDA], kbA);
#pragma unroll
    for (int nt = 0; nt < 4; ++nt) {
      v16h bF = loadB(&lB[(wn * 64 + nt * 16 + mr) * LDB], kbB);
      acc[0][nt] = wmma_f16(aF[0], bF, acc[0][nt]);
      acc[1][nt] = wmma_f16(aF[1], bF, acc[1][nt]);
    }
    __syncthreads();
  }

  // Epilogue: n -> (head h, hd, which) per interleaved [h][hd][3] layout.
  const int hiRow = (lane < 16) ? 0 : 8;
#pragma unroll
  for (int mt = 0; mt < 2; ++mt)
#pragma unroll
    for (int nt = 0; nt < 4; ++nt) {
      int n = nBase + wn * 64 + nt * 16 + mr;
      int h = n / 192, rem = n % 192;
      int hd = rem / 3, which = rem % 3;
#pragma unroll
      for (int r = 0; r < 8; ++r) {
        int m = mBase + wm * 32 + mt * 16 + r + hiRow;
        int b = m >> 11, s = m & 2047;
        int bh = b * 16 + h;
        float v = acc[mt][nt][r];
        if (which == 0)
          Qh[((size_t)bh * 2048 + s) * 64 + hd] = (_Float16)(v * 0.03125f);
        else if (which == 1)
          Kh[((size_t)bh * 2048 + s) * 64 + hd] = (_Float16)v;
        else
          Vt[((size_t)bh * 64 + hd) * 2048 + s] = (_Float16)v;
      }
    }
}

// ---------------------------------------------------------------------------
// Kernel 2: flash attention.  grid (16 qblocks, 64 bh), block 256.
// Each wave: 16 q-rows; key loop over 2048 in chunks of 64.
// ---------------------------------------------------------------------------
__global__ __launch_bounds__(256) void attn_kernel(
    const _Float16* __restrict__ Qh, const _Float16* __restrict__ Kh,
    const _Float16* __restrict__ Vt, _Float16* __restrict__ Attn)
{
  constexpr int LDK = 72, LDV = 72, LDP = 72;
  __shared__ _Float16 lK[64 * LDK];
  __shared__ _Float16 lV[64 * LDV];
  __shared__ _Float16 lP[8 * 16 * LDP];

  const int tid  = threadIdx.x;
  const int lane = tid & 31, wave = tid >> 5;
  const int bh    = blockIdx.y;
  const int qBase = blockIdx.x * 128;
  const size_t bhOff = (size_t)bh * 2048 * 64;

  const int kbA = (lane < 16) ? 0 : 16;
  const int kbB = (lane < 16) ? 0 : 16;
  const int mr  = lane & 15;
  const int hiRow = (lane < 16) ? 0 : 8;

  // Q fragments for this wave's 16 rows (scale already folded in).
  v16h qF[2];
#pragma unroll
  for (int c = 0; c < 2; ++c)
    qF[c] = loadA(Qh + bhOff + (size_t)(qBase + wave * 16 + mr) * 64 + c * 32, kbA);

  float mrow[8], lrow[8];
  v8f O[4];
#pragma unroll
  for (int r = 0; r < 8; ++r) { mrow[r] = -1e30f; lrow[r] = 0.f; }
#pragma unroll
  for (int j = 0; j < 4; ++j)
    O[j] = (v8f){0.f,0.f,0.f,0.f,0.f,0.f,0.f,0.f};

  _Float16* myP = &lP[wave * 16 * LDP];

  for (int kc = 0; kc < 2048; kc += 64) {
    // K tile (64 keys x 64 d) -> lK[key][d];  V^T tile (64 d x 64 k) -> lV[d][key]
#if USE_ASYNC_LDS
#pragma unroll
    for (int i = 0; i < 2; ++i) {
      int idx = tid + i * 256;               // 512 b128 copies per tile
      int row = idx >> 3, cv = (idx & 7) * 8;
      ASYNC_CP16(Kh + bhOff + (size_t)(kc + row) * 64 + cv, &lK[row * LDK + cv]);
      ASYNC_CP16(Vt + bhOff + (size_t)row * 2048 + kc + cv, &lV[row * LDV + cv]);
    }
    ASYNC_WAIT();
#else
#pragma unroll
    for (int i = 0; i < 2; ++i) {
      int idx = tid + i * 256;
      int row = idx >> 3, cv = (idx & 7) * 8;
      *(v8h*)&lK[row * LDK + cv] =
          *(const v8h*)(Kh + bhOff + (size_t)(kc + row) * 64 + cv);
      *(v8h*)&lV[row * LDV + cv] =
          *(const v8h*)(Vt + bhOff + (size_t)row * 2048 + kc + cv);
    }
#endif
    __syncthreads();

    // S = Q @ K^T for this 16x64 block.
    v8f S[4];
#pragma unroll
    for (int j = 0; j < 4; ++j) {
      S[j] = (v8f){0.f,0.f,0.f,0.f,0.f,0.f,0.f,0.f};
#pragma unroll
      for (int c = 0; c < 2; ++c) {
        v16h bF = loadB(&lK[(j * 16 + mr) * LDK + c * 32], kbB);
        S[j] = wmma_f16(qF[c], bF, S[j]);
      }
    }

    // Online softmax update.
    float cm[8];
#pragma unroll
    for (int r = 0; r < 8; ++r) {
      float v = S[0][r];
      v = fmaxf(v, S[1][r]); v = fmaxf(v, S[2][r]); v = fmaxf(v, S[3][r]);
      v = rmax16(v);
      float mn = fmaxf(mrow[r], v);
      cm[r] = __expf(mrow[r] - mn);          // rescale factor
      mrow[r] = mn;
    }
    float rs[8];
#pragma unroll
    for (int r = 0; r < 8; ++r) rs[r] = 0.f;
#pragma unroll
    for (int j = 0; j < 4; ++j)
#pragma unroll
      for (int r = 0; r < 8; ++r) {
        float p = __expf(S[j][r] - mrow[r]);
        S[j][r] = p;
        rs[r] += p;
      }
#pragma unroll
    for (int r = 0; r < 8; ++r)
      lrow[r] = lrow[r] * cm[r] + rsum16(rs[r]);
#pragma unroll
    for (int j = 0; j < 4; ++j)
#pragma unroll
      for (int r = 0; r < 8; ++r)
        O[j][r] *= cm[r];

    // P (C-layout) -> per-wave LDS -> A-layout fragments.
#pragma unroll
    for (int j = 0; j < 4; ++j)
#pragma unroll
      for (int r = 0; r < 8; ++r)
        myP[(r + hiRow) * LDP + j * 16 + mr] = (_Float16)S[j][r];

    v16h pF[2];
#pragma unroll
    for (int c = 0; c < 2; ++c)
      pF[c] = loadA(&myP[mr * LDP + c * 32], kbA);

    // O += P @ V
#pragma unroll
    for (int j = 0; j < 4; ++j)
#pragma unroll
      for (int c = 0; c < 2; ++c) {
        v16h bF = loadB(&lV[(j * 16 + mr) * LDV + c * 32], kbB);
        O[j] = wmma_f16(pF[c], bF, O[j]);
      }
    __syncthreads();
  }

  // Finalize: O /= l, write Attn[token][h*64+d] as f16.
  const int b = bh >> 4, h = bh & 15;
#pragma unroll
  for (int j = 0; j < 4; ++j)
#pragma unroll
    for (int r = 0; r < 8; ++r) {
      int srow = qBase + wave * 16 + r + hiRow;
      size_t tok = (size_t)b * 2048 + srow;
      Attn[tok * 1024 + h * 64 + j * 16 + mr] = (_Float16)(O[j][r] / lrow[r]);
    }
}

// ---------------------------------------------------------------------------
// Kernel 3: out = Attn(f16) @ W_Out.  grid (8, 64), block 256.
// ---------------------------------------------------------------------------
__global__ __launch_bounds__(256) void out_kernel(
    const _Float16* __restrict__ Attn, const float* __restrict__ Wout,
    float* __restrict__ out)
{
  constexpr int LDA = 40, LDB = 40;
  __shared__ _Float16 lA[128 * LDA];
  __shared__ _Float16 lB[128 * LDB];

  const int tid  = threadIdx.x;
  const int lane = tid & 31, wave = tid >> 5;
  const int wm = wave >> 1, wn = wave & 1;
  const int mBase = blockIdx.y * 128;
  const int nBase = blockIdx.x * 128;

  v8f acc[2][4];
#pragma unroll
  for (int i = 0; i < 2; ++i)
#pragma unroll
    for (int j = 0; j < 4; ++j)
      acc[i][j] = (v8f){0.f,0.f,0.f,0.f,0.f,0.f,0.f,0.f};

  const int kbA = (lane < 16) ? 0 : 16;
  const int kbB = (lane < 16) ? 0 : 16;
  const int mr  = lane & 15;

  for (int k0 = 0; k0 < 1024; k0 += 32) {
    // A tile: Attn[mBase..+127][k0..k0+31] (already f16), [m][k]
#if USE_ASYNC_LDS
#pragma unroll
    for (int i = 0; i < 2; ++i) {
      int idx = tid + i * 256;               // 512 b128 copies
      int row = idx >> 2, cv = (idx & 3) * 8;
      ASYNC_CP16(Attn + (size_t)(mBase + row) * 1024 + k0 + cv,
                 &lA[row * LDA + cv]);
    }
#else
#pragma unroll
    for (int i = 0; i < 2; ++i) {
      int idx = tid + i * 256;
      int row = idx >> 2, cv = (idx & 3) * 8;
      *(v8h*)&lA[row * LDA + cv] =
          *(const v8h*)(Attn + (size_t)(mBase + row) * 1024 + k0 + cv);
    }
#endif
    // B tile: Wout[k0..+31][nBase..+127] f32 -> f16, transposed to [n][k]
    // (VALU conversion overlaps the async A-tile transfer.)
#pragma unroll
    for (int i = 0; i < 4; ++i) {
      int idx = tid + i * 256;
      int r = idx >> 5, nv = (idx & 31) * 4;
      v4f d = *(const v4f*)(Wout + (size_t)(k0 + r) * 1024 + nBase + nv);
      lB[(nv + 0) * LDB + r] = (_Float16)d.x;
      lB[(nv + 1) * LDB + r] = (_Float16)d.y;
      lB[(nv + 2) * LDB + r] = (_Float16)d.z;
      lB[(nv + 3) * LDB + r] = (_Float16)d.w;
    }
#if USE_ASYNC_LDS
    ASYNC_WAIT();
#endif
    __syncthreads();

    v16h aF[2];
#pragma unroll
    for (int mt = 0; mt < 2; ++mt)
      aF[mt] = loadA(&lA[(wm * 32 + mt * 16 + mr) * LDA], kbA);
#pragma unroll
    for (int nt = 0; nt < 4; ++nt) {
      v16h bF = loadB(&lB[(wn * 64 + nt * 16 + mr) * LDB], kbB);
      acc[0][nt] = wmma_f16(aF[0], bF, acc[0][nt]);
      acc[1][nt] = wmma_f16(aF[1], bF, acc[1][nt]);
    }
    __syncthreads();
  }

  const int hiRow = (lane < 16) ? 0 : 8;
#pragma unroll
  for (int mt = 0; mt < 2; ++mt)
#pragma unroll
    for (int nt = 0; nt < 4; ++nt) {
      int n = nBase + wn * 64 + nt * 16 + mr;
#pragma unroll
      for (int r = 0; r < 8; ++r) {
        int m = mBase + wm * 32 + mt * 16 + r + hiRow;
        out[(size_t)m * 1024 + n] = acc[mt][nt][r];
      }
    }
}

// ---------------------------------------------------------------------------
extern "C" void kernel_launch(void* const* d_in, const int* in_sizes, int n_in,
                              void* d_out, int out_size, void* d_ws, size_t ws_size,
                              hipStream_t stream) {
  (void)in_sizes; (void)n_in; (void)out_size; (void)ws_size;
  const float* x    = (const float*)d_in[0];
  const float* Wqkv = (const float*)d_in[1];
  const float* Wout = (const float*)d_in[2];
  float* out = (float*)d_out;

  char* ws = (char*)d_ws;                     // needs 64 MB scratch
  _Float16* Qh = (_Float16*)(ws + (size_t)0 * (16u << 20));
  _Float16* Kh = (_Float16*)(ws + (size_t)1 * (16u << 20));
  _Float16* Vt = (_Float16*)(ws + (size_t)2 * (16u << 20));
  _Float16* At = (_Float16*)(ws + (size_t)3 * (16u << 20));

  qkv_kernel <<<dim3(24, 64), 256, 0, stream>>>(x, Wqkv, Qh, Kh, Vt);
  attn_kernel<<<dim3(16, 64), 256, 0, stream>>>(Qh, Kh, Vt, At);
  out_kernel <<<dim3(8, 64), 256, 0, stream>>>(At, Wout, out);
}
